// SpatialLocalAttention_26774826123569
// MI455X (gfx1250) — matrix operand order repaired and checked by reading, not compiled, not run on hardware
//
#include <hip/hip_runtime.h>
#include <hip/hip_bf16.h>
#include <math.h>

#define B_     2
#define L_     2048
#define DIM_   256
#define PE_    32
#define HEADS_ 8
#define DHEAD_ 64
#define INNER_ 512
#define KNBR_  32
#define G_     8
#define CTX_   41   // 1 self + 32 neighbors + 8 globals

typedef __attribute__((ext_vector_type(16))) _Float16 v16h;
typedef __attribute__((ext_vector_type(2)))  _Float16 v2h;
typedef __attribute__((ext_vector_type(8)))  float    v8f;

// ---------------------------------------------------------------------------
// Generic GEMM:  C[M,N] = A[M,K] @ W[K,N] (+ bias[N]),  fp32 in/out,
// f16 WMMA (v_wmma_f32_16x16x32_f16) with fp32 accumulation.
// Tile: BM=64, BN=128, BK=32. 256 threads = 8 wave32 waves.
// Wave w computes rows (w&3)*16 .. +15, cols (w>>2)*64 .. +63  (4 C frags).
//
// LDS tiles stored in WMMA fragment order (ISA 7.12.2 per-lane layout):
//   A subtile (16x32): elem(m,K) -> lane m + (K&8?16:0), slot (K&7)|((K>>1)&8)
//   B subtile (32x16): elem(K,n) -> lane (n&15) + (K&8?16:0), slot (K&7)|((K>>1)&8)
// so each operand fetch is one contiguous 32 B per-lane read (2x ds_load_b128).
//
// Software-pipelined: double-buffered LDS; global loads for step k+1 are all
// issued (batched, branch-free) before the WMMAs of step k, converted f32->f16
// and staged afterwards. One barrier per k-step.
// ---------------------------------------------------------------------------
__global__ __launch_bounds__(256)
void gemm_wmma_f16(const float* __restrict__ A, int lda,
                   const float* __restrict__ W, int ldw,
                   float* __restrict__ C,
                   int M, int K, int N,
                   const float* __restrict__ bias)
{
  __shared__ __align__(32) _Float16 As[2][4 * 512];
  __shared__ __align__(32) _Float16 Bs[2][8 * 512];

  const int tid    = threadIdx.x;
  const int lane   = tid & 31;
  const int wid    = tid >> 5;
  const int waveM  = (wid & 3) * 16;      // row offset in 64-row tile
  const int waveN  = (wid >> 2) * 64;     // col offset in 128-col tile
  const int blockM = blockIdx.x * 64;
  const int blockN = blockIdx.y * 128;

  // ---- Loop-invariant staging descriptors --------------------------------
  // A: 4 K-pairs per thread (64x32 tile = 1024 pairs / 256 thr).
  const float* aptr[4]; int adst[4]; float amask[4];
#pragma unroll
  for (int i = 0; i < 4; ++i) {
    int e  = tid + i * 256;
    int r  = e >> 4;                      // row in tile 0..63
    int kp = (e & 15) << 1;               // even K 0..30
    int gr = blockM + r;
    amask[i] = (gr < M) ? 1.0f : 0.0f;    // branch-free OOB handling
    int grc  = (gr < M) ? gr : (M - 1);   // clamped, always-valid address
    aptr[i]  = A + (size_t)grc * lda + kp;
    int sub   = r >> 4;
    int laned = (r & 15) + ((kp & 8) ? 16 : 0);
    int slot  = (kp & 7) | ((kp >> 1) & 8);
    adst[i]   = sub * 512 + laned * 16 + slot;
  }
  // B: 8 K-pairs per thread (32x128 tile = 2048 pairs / 256 thr).
  const float* bptr[8]; int bdst[8];
#pragma unroll
  for (int i = 0; i < 8; ++i) {
    int e  = tid + i * 256;
    int nf = e & 127;                     // col in tile
    int kp = (e >> 7) << 1;               // even K 0..30
    bptr[i] = W + (size_t)kp * ldw + blockN + nf;
    int nt    = nf >> 4;
    int laned = (nf & 15) + ((kp & 8) ? 16 : 0);
    int slot  = (kp & 7) | ((kp >> 1) & 8);
    bdst[i]   = nt * 512 + laned * 16 + slot;
  }

  v8f acc[4] = {};
  float2 areg[4];
  float  breg0[8], breg1[8];

  // Batched global loads for k-step ks (no intervening waits).
  auto gload = [&](int ks) {
    const size_t ao = (size_t)ks << 5;
#pragma unroll
    for (int i = 0; i < 4; ++i) areg[i] = *(const float2*)(aptr[i] + ao);
    const size_t bo = ((size_t)ks << 5) * (size_t)ldw;
#pragma unroll
    for (int i = 0; i < 8; ++i) {
      breg0[i] = bptr[i][bo];
      breg1[i] = bptr[i][bo + ldw];
    }
  };
  // Convert + store staged registers into LDS buffer `buf`.
  auto stage = [&](int buf) {
#pragma unroll
    for (int i = 0; i < 4; ++i) {
      *(v2h*)&As[buf][adst[i]] =
          (v2h){(_Float16)(areg[i].x * amask[i]), (_Float16)(areg[i].y * amask[i])};
    }
#pragma unroll
    for (int i = 0; i < 8; ++i) {
      *(v2h*)&Bs[buf][bdst[i]] = (v2h){(_Float16)breg0[i], (_Float16)breg1[i]};
    }
  };

  const int ksteps = K >> 5;
  gload(0);
  stage(0);
  __syncthreads();

  for (int ks = 0; ks < ksteps; ++ks) {
    const int cur = ks & 1;
    if (ks + 1 < ksteps) gload(ks + 1);   // issue next slab early

    v16h afrag = *(const v16h*)&As[cur][(waveM >> 4) * 512 + lane * 16];
#pragma unroll
    for (int nt = 0; nt < 4; ++nt) {
      v16h bfrag = *(const v16h*)&Bs[cur][((waveN >> 4) + nt) * 512 + lane * 16];
      acc[nt] = __builtin_amdgcn_wmma_f32_16x16x32_f16(
          false, afrag, false, bfrag, (short)0, acc[nt], false, false);
    }
    if (ks + 1 < ksteps) stage(cur ^ 1);  // fill the other buffer
    __syncthreads();
  }

  // C layout: VGPR r -> M = r (lanes 0-15) or r+8 (lanes 16-31), N = lane&15.
  const int rowhi = (lane >> 4) << 3;
#pragma unroll
  for (int nt = 0; nt < 4; ++nt) {
#pragma unroll
    for (int r = 0; r < 8; ++r) {
      int grow = blockM + waveM + r + rowhi;
      int gcol = blockN + waveN + nt * 16 + (lane & 15);
      if (grow < M) {
        float o = acc[nt][r];
        if (bias) o += bias[gcol];
        C[(size_t)grow * N + gcol] = o;
      }
    }
  }
}

// ---------------------------------------------------------------------------
// Fused gather-attention. One block (256 thr = 8 waves) per query (b,l);
// wave h handles head h. Uses precomputed Q, Sk=spatial@Wk[:256],
// Sv=spatial@Wv[:256], Gk/Gv=latents@W[:256].
// rpe folds in via qtil = Wkr_h @ q_h  (scores) and rbar @ Wvr_h (values).
// ---------------------------------------------------------------------------
__global__ __launch_bounds__(256)
void attn_fused(const float* __restrict__ Q,
                const float* __restrict__ Sk,
                const float* __restrict__ Sv,
                const float* __restrict__ Gk,
                const float* __restrict__ Gv,
                const int*   __restrict__ topk,
                const float* __restrict__ rpe,
                const float* __restrict__ self_rpe,
                const float* __restrict__ dist,
                const float* __restrict__ Wk,
                const float* __restrict__ Wv,
                const float* __restrict__ log_sigma,
                const float* __restrict__ global_bias,
                float* __restrict__ AVout)
{
  __shared__ float s_q[INNER_];
  __shared__ float s_rpe[33][PE_];
  __shared__ float s_qtil[HEADS_][PE_];
  __shared__ float s_w[HEADS_][CTX_];
  __shared__ int   s_idx[KNBR_];
  __shared__ float s_dist[KNBR_];

  const int tid  = threadIdx.x;
  const int lane = tid & 31;
  const int h    = tid >> 5;
  const size_t bl   = blockIdx.x;       // b*L + l
  const size_t qoff = bl * INNER_;
  const size_t b    = bl / L_;

  s_q[tid]       = Q[qoff + tid];
  s_q[tid + 256] = Q[qoff + tid + 256];
  for (int e = tid; e < 33 * PE_; e += 256) {
    int c = e >> 5, p = e & 31;
    s_rpe[c][p] = (c == 0)
        ? self_rpe[bl * PE_ + p]
        : rpe[bl * (size_t)(KNBR_ * PE_) + (size_t)(c - 1) * PE_ + p];
  }
  if (tid < KNBR_) {
    s_idx[tid]  = topk[bl * KNBR_ + tid];
    s_dist[tid] = dist[bl * KNBR_ + tid];
  }
  __syncthreads();

  // qtil[h][p] = sum_d q[h,d] * Wk[256+p, h*64+d]   (lane p)
  {
    const float* wkr = Wk + (size_t)(256 + lane) * INNER_ + h * DHEAD_;
    const float* qh  = s_q + h * DHEAD_;
    float s = 0.f;
#pragma unroll 8
    for (int d = 0; d < DHEAD_; ++d) s += qh[d] * wkr[d];
    s_qtil[h][lane] = s;
  }
  __syncthreads();

  const float sg    = expf(log_sigma[h]);
  const float inv2s = 1.0f / (2.0f * sg * sg);
  const float gb    = global_bias[0];

  // Raw scores: each lane covers ctx c = lane and lane+32 (if < 41).
  float raw[2] = {0.f, 0.f};
#pragma unroll
  for (int t = 0; t < 2; ++t) {
    int c = lane + t * 32;
    if (c < CTX_) {
      const float* key;
      float biasv;
      if (c == 0) {
        key = Sk + qoff + h * DHEAD_;                      biasv = 0.f;
      } else if (c <= KNBR_) {
        key = Sk + (b * L_ + (size_t)s_idx[c - 1]) * INNER_ + h * DHEAD_;
        float dd = s_dist[c - 1];                          biasv = -dd * dd * inv2s;
      } else {
        key = Gk + (b * G_ + (size_t)(c - 33)) * INNER_ + h * DHEAD_;
        biasv = gb;
      }
      const float* qh = s_q + h * DHEAD_;
      float s = 0.f;
#pragma unroll 8
      for (int d = 0; d < DHEAD_; ++d) s += qh[d] * key[d];
      if (c < 33) {                                  // rpe term (globals: zero rpe)
#pragma unroll
        for (int p = 0; p < PE_; ++p) s += s_qtil[h][p] * s_rpe[c][p];
      }
      raw[t] = 0.125f * s + biasv;                   // scale = DIM_HEAD^-0.5
      s_w[h][c] = raw[t];
    }
  }
  __syncthreads();

  // Softmax over 41 (lockstep wave: all reads precede any write).
  float mx = -1e30f;
  for (int c = 0; c < CTX_; ++c) mx = fmaxf(mx, s_w[h][c]);
  float sum = 0.f;
  for (int c = 0; c < CTX_; ++c) sum += expf(s_w[h][c] - mx);
  float inv = 1.0f / sum;
#pragma unroll
  for (int t = 0; t < 2; ++t) {
    int c = lane + t * 32;
    if (c < CTX_) s_w[h][c] = expf(raw[t] - mx) * inv;
  }
  __syncthreads();

  // Weighted value sum: lane covers d = lane and lane+32 (coalesced gathers).
  float acc0 = 0.f, acc1 = 0.f;
  for (int c = 0; c < CTX_; ++c) {
    float w = s_w[h][c];
    const float* vrow;
    if (c == 0)           vrow = Sv + qoff + h * DHEAD_;
    else if (c <= KNBR_)  vrow = Sv + (b * L_ + (size_t)s_idx[c - 1]) * INNER_ + h * DHEAD_;
    else                  vrow = Gv + (b * G_ + (size_t)(c - 33)) * INNER_ + h * DHEAD_;
    acc0 += w * vrow[lane];
    acc1 += w * vrow[lane + 32];
  }
  // rbar[p] = sum_{c<33} w_c * rpe_c[p]   (lane p)
  float rb = 0.f;
#pragma unroll 8
  for (int c = 0; c < 33; ++c) rb += s_w[h][c] * s_rpe[c][lane];
  // out += rbar @ Wv[256:, h-block]
#pragma unroll 4
  for (int p = 0; p < PE_; ++p) {
    float rp = __shfl(rb, p, 32);
    const float* wvr = Wv + (size_t)(256 + p) * INNER_ + h * DHEAD_;
    acc0 += rp * wvr[lane];
    acc1 += rp * wvr[lane + 32];
  }
  AVout[qoff + h * DHEAD_ + lane]      = acc0;
  AVout[qoff + h * DHEAD_ + lane + 32] = acc1;
}

// ---------------------------------------------------------------------------
extern "C" void kernel_launch(void* const* d_in, const int* in_sizes, int n_in,
                              void* d_out, int out_size, void* d_ws, size_t ws_size,
                              hipStream_t stream) {
  const float* spatial   = (const float*)d_in[0];
  const int*   topk      = (const int*)  d_in[1];
  const float* rpe       = (const float*)d_in[2];
  const float* self_rpe  = (const float*)d_in[3];
  const float* dist      = (const float*)d_in[4];
  const float* glob      = (const float*)d_in[5];
  const float* Wq        = (const float*)d_in[6];
  const float* Wk        = (const float*)d_in[7];
  const float* Wv        = (const float*)d_in[8];
  const float* Wo        = (const float*)d_in[9];
  const float* bo        = (const float*)d_in[10];
  const float* log_sigma = (const float*)d_in[11];
  const float* gbias     = (const float*)d_in[12];
  float* out = (float*)d_out;

  const size_t NQ = (size_t)B_ * L_;       // 4096 queries
  float* ws = (float*)d_ws;
  float* q  = ws;                          // NQ*512
  float* Sk = q  + NQ * INNER_;            // NQ*512
  float* Sv = Sk + NQ * INNER_;            // NQ*512
  float* AV = Sv + NQ * INNER_;            // NQ*512
  float* Gk = AV + NQ * INNER_;            // 16*512
  float* Gv = Gk + (size_t)B_ * G_ * INNER_;
  // total ws use: ~33.6 MB

  dim3 blk(256);
  // Projections of unique feature rows (f16 WMMA, fp32 accum).
  gemm_wmma_f16<<<dim3((unsigned)(NQ / 64), INNER_ / 128), blk, 0, stream>>>(
      spatial, DIM_, Wq, INNER_, q,  (int)NQ, DIM_, INNER_, nullptr);
  gemm_wmma_f16<<<dim3((unsigned)(NQ / 64), INNER_ / 128), blk, 0, stream>>>(
      spatial, DIM_, Wk, INNER_, Sk, (int)NQ, DIM_, INNER_, nullptr);
  gemm_wmma_f16<<<dim3((unsigned)(NQ / 64), INNER_ / 128), blk, 0, stream>>>(
      spatial, DIM_, Wv, INNER_, Sv, (int)NQ, DIM_, INNER_, nullptr);
  gemm_wmma_f16<<<dim3(1, INNER_ / 128), blk, 0, stream>>>(
      glob, DIM_, Wk, INNER_, Gk, B_ * G_, DIM_, INNER_, nullptr);
  gemm_wmma_f16<<<dim3(1, INNER_ / 128), blk, 0, stream>>>(
      glob, DIM_, Wv, INNER_, Gv, B_ * G_, DIM_, INNER_, nullptr);

  // Fused gather-attention (L2-resident gathers).
  attn_fused<<<dim3((unsigned)NQ), blk, 0, stream>>>(
      q, Sk, Sv, Gk, Gv, topk, rpe, self_rpe, dist, Wk, Wv,
      log_sigma, gbias, AV);

  // Output projection + bias.
  gemm_wmma_f16<<<dim3((unsigned)(NQ / 64), DIM_ / 128), blk, 0, stream>>>(
      AV, INNER_, Wo, DIM_, out, (int)NQ, INNER_, DIM_, bo);
}
